// BertSelfAttention_91250875170938
// MI455X (gfx1250) — matrix operand order repaired and checked by reading, not compiled, not run on hardware
//
#include <hip/hip_runtime.h>

#define HIDDEN 1024
#define HEADS  16
#define HDIM   64
#define NB     4
#define SEQ    2048
#define MTOT   (NB*SEQ)   // 8192

typedef __attribute__((ext_vector_type(16))) _Float16 v16h;
typedef __attribute__((ext_vector_type(8)))  _Float16 v8h;
typedef __attribute__((ext_vector_type(4)))  _Float16 v4h;
typedef __attribute__((ext_vector_type(8)))  float    v8f;
typedef __attribute__((ext_vector_type(4)))  float    v4f;
typedef __attribute__((ext_vector_type(4)))  int      v4i;

// ---- CDNA5 async global->LDS copy (ASYNCcnt-tracked), with safe fallback ----
#if defined(__gfx1250__) && __has_builtin(__builtin_amdgcn_global_load_async_to_lds_b128) && \
    __has_builtin(__builtin_amdgcn_s_wait_asynccnt)
#define HAVE_ASYNC_LDS 1
#else
#define HAVE_ASYNC_LDS 0
#endif

#if HAVE_ASYNC_LDS
typedef __attribute__((address_space(1))) v4i gas_v4i;
typedef __attribute__((address_space(3))) v4i las_v4i;
#endif

static __device__ __forceinline__ void async_cp16(const _Float16* g, _Float16* l) {
#if HAVE_ASYNC_LDS
  __builtin_amdgcn_global_load_async_to_lds_b128(
      (gas_v4i*)(size_t)g, (las_v4i*)(unsigned)(size_t)l, 0, 0);
#else
  *(v8h*)l = *(const v8h*)g;
#endif
}

static __device__ __forceinline__ v16h cat8(v8h lo, v8h hi) {
  union { v16h v; v8h h[2]; } u; u.h[0] = lo; u.h[1] = hi; return u.v;
}

// A-operand (16x32 f16): lane holds row (rowBase + lane%16);
// halves k = koff + 8*(lane/16) + {0..7} and koff + 16 + 8*(lane/16) + {0..7}
static __device__ __forceinline__ v16h opA(const _Float16* buf, int stride,
                                           int rowBase, int koff, int lane) {
  const _Float16* p = buf + (rowBase + (lane & 15)) * stride + koff + ((lane >> 4) << 3);
  return cat8(*(const v8h*)p, *(const v8h*)(p + 16));
}

// B-operand (32x16 f16, stored as B^T rows): lane holds row (rowBase + lane%16);
// halves k = koff + 8*(lane/16) + {0..7} and koff + 16 + 8*(lane/16) + {0..7}
static __device__ __forceinline__ v16h opB(const _Float16* buf, int stride,
                                           int rowBase, int koff, int lane) {
  const _Float16* p = buf + (rowBase + (lane & 15)) * stride + koff + ((lane >> 4) << 3);
  return cat8(*(const v8h*)p, *(const v8h*)(p + 16));
}

#define WMMA(A, B, C) \
  __builtin_amdgcn_wmma_f32_16x16x32_f16(false, (A), false, (B), (short)0, (C), false, false)

// ---------------------------------------------------------------------------
// Pre-pass 1: X fp32 -> f16, elementwise.
// ---------------------------------------------------------------------------
__global__ __launch_bounds__(256) void cvt_x_kernel(const float* __restrict__ x,
                                                    _Float16* __restrict__ xh) {
  size_t i = ((size_t)blockIdx.x * 256 + threadIdx.x) * 8;
  v4f a = *(const v4f*)(x + i);
  v4f b = *(const v4f*)(x + i + 4);
  v8h o;
#pragma unroll
  for (int r = 0; r < 4; r++) { o[r] = (_Float16)a[r]; o[r + 4] = (_Float16)b[r]; }
  *(v8h*)(xh + i) = o;
}

// ---------------------------------------------------------------------------
// Pre-pass 2: W [k][n] fp32 -> Wt [n][k] f16, 32x32 LDS tile transpose.
// ---------------------------------------------------------------------------
__global__ __launch_bounds__(256) void cvt_wt_kernel(const float* __restrict__ W0,
                                                     const float* __restrict__ W1,
                                                     const float* __restrict__ W2,
                                                     _Float16* __restrict__ Wt) {
  const float* W = (blockIdx.z == 0) ? W0 : ((blockIdx.z == 1) ? W1 : W2);
  _Float16* out = Wt + (size_t)blockIdx.z * HIDDEN * HIDDEN;
  __shared__ _Float16 tile[32][33];
  int k0 = blockIdx.y * 32, n0 = blockIdx.x * 32;
  int r = threadIdx.x >> 3, c4 = (threadIdx.x & 7) * 4;
  v4f a = *(const v4f*)(W + (size_t)(k0 + r) * HIDDEN + n0 + c4);
#pragma unroll
  for (int i = 0; i < 4; i++) tile[c4 + i][r] = (_Float16)a[i];
  __syncthreads();
  v4h o;
#pragma unroll
  for (int i = 0; i < 4; i++) o[i] = tile[r][c4 + i];
  *(v4h*)(out + (size_t)(n0 + r) * HIDDEN + k0 + c4) = o;
}

// ---------------------------------------------------------------------------
// Fused QKV projection GEMM, double-buffered async staging.
// Block tile 256(M) x 128(N), K-step 32, 8 waves each owning 64x64.
// QK=true : D[n][m] (A=Wt, B=X) -> [b,h,s,d] (Q folds 1/8); QK=false: D[m][n] -> [b,h,d,s]
// ---------------------------------------------------------------------------
static __device__ __forceinline__ void stage_proj(const _Float16* Xh, const _Float16* Wt,
                                                  _Float16* xbuf, _Float16* wbuf,
                                                  int m0, int n0, int k0, int tid) {
#pragma unroll
  for (int j = 0; j < 4; j++) {  // X tile: 256 rows x 32 halves
    int c = tid + j * 256;
    int row = c >> 2, col = (c & 3) << 3;
    async_cp16(Xh + (size_t)(m0 + row) * HIDDEN + k0 + col, xbuf + row * 40 + col);
  }
#pragma unroll
  for (int j = 0; j < 2; j++) {  // Wt tile: 128 rows x 32 halves
    int c = tid + j * 256;
    int row = c >> 2, col = (c & 3) << 3;
    async_cp16(Wt + (size_t)(n0 + row) * HIDDEN + k0 + col, wbuf + row * 40 + col);
  }
}

template <bool QK>
__global__ __launch_bounds__(256) void proj_kernel(const _Float16* __restrict__ Xh,
                                                   const _Float16* __restrict__ Wt,
                                                   const float* __restrict__ bias,
                                                   _Float16* __restrict__ outbuf,
                                                   float outScale) {
  constexpr int BM = 256, BN = 128, LSTR = 40;  // 32 + 8 pad halves
  __shared__ alignas(16) _Float16 xs[2][BM * LSTR];
  __shared__ alignas(16) _Float16 wsh[2][BN * LSTR];

  int n0 = blockIdx.x * BN, m0 = blockIdx.y * BM;
  int tid = threadIdx.x, lane = tid & 31, w = tid >> 5;
  int wm = (w & 3) * 64, wn = (w >> 2) * 64;

  v8f c[4][4];
  v8f z8 = {0.f, 0.f, 0.f, 0.f, 0.f, 0.f, 0.f, 0.f};
#pragma unroll
  for (int a = 0; a < 4; a++)
#pragma unroll
    for (int b = 0; b < 4; b++) c[a][b] = z8;

  stage_proj(Xh, Wt, xs[0], wsh[0], m0, n0, 0, tid);

  for (int k0 = 0; k0 < HIDDEN; k0 += 32) {
    int cur = (k0 >> 5) & 1;
    bool haveNext = (k0 + 32) < HIDDEN;
    if (haveNext) stage_proj(Xh, Wt, xs[cur ^ 1], wsh[cur ^ 1], m0, n0, k0 + 32, tid);
#if HAVE_ASYNC_LDS
    if (haveNext) __builtin_amdgcn_s_wait_asynccnt(6);  // leave next batch in flight
    else          __builtin_amdgcn_s_wait_asynccnt(0);
#endif
    __syncthreads();

    const _Float16* xb = xs[cur];
    const _Float16* wb = wsh[cur];
    v16h A[4], Bv[4];
#pragma unroll
    for (int a = 0; a < 4; a++)
      A[a] = QK ? opA(wb, LSTR, wn + a * 16, 0, lane) : opA(xb, LSTR, wm + a * 16, 0, lane);
#pragma unroll
    for (int b = 0; b < 4; b++)
      Bv[b] = QK ? opB(xb, LSTR, wm + b * 16, 0, lane) : opB(wb, LSTR, wn + b * 16, 0, lane);
#pragma unroll
    for (int a = 0; a < 4; a++)
#pragma unroll
      for (int b = 0; b < 4; b++) c[a][b] = WMMA(A[a], Bv[b], c[a][b]);
    __syncthreads();
  }

  int roff = (lane >> 4) << 3;  // 0 or 8
  if (QK) {
#pragma unroll
    for (int a = 0; a < 4; a++) {
      int nbase = n0 + wn + a * 16 + roff;
      int h = nbase >> 6, dbase = nbase & 63;
      v4f bA = *(const v4f*)(bias + nbase);
      v4f bB = *(const v4f*)(bias + nbase + 4);
#pragma unroll
      for (int b = 0; b < 4; b++) {
        int mcol = m0 + wm + b * 16 + (lane & 15);
        int bidx = mcol >> 11, s = mcol & (SEQ - 1);
        v8h o;
#pragma unroll
        for (int r = 0; r < 4; r++) {
          o[r] = (_Float16)((c[a][b][r] + bA[r]) * outScale);
          o[r + 4] = (_Float16)((c[a][b][r + 4] + bB[r]) * outScale);
        }
        size_t addr = (((size_t)bidx * HEADS + h) * SEQ + s) * HDIM + dbase;
        *(v8h*)(outbuf + addr) = o;
      }
    }
  } else {
#pragma unroll
    for (int b = 0; b < 4; b++) {
      int ncol = n0 + wn + b * 16 + (lane & 15);
      int h = ncol >> 6, d = ncol & 63;
      float bv = bias[ncol];
#pragma unroll
      for (int a = 0; a < 4; a++) {
        int mbase = m0 + wm + a * 16 + roff;
        int bidx = mbase >> 11, s = mbase & (SEQ - 1);
        v8h o;
#pragma unroll
        for (int r = 0; r < 8; r++) o[r] = (_Float16)((c[a][b][r] + bv) * outScale);
        size_t addr = (((size_t)bidx * HEADS + h) * HDIM + d) * SEQ + s;
        *(v8h*)(outbuf + addr) = o;
      }
    }
  }
}

// ---------------------------------------------------------------------------
// Flash attention, 64-key steps, double-buffered async K/V staging.
// Block = one (b,h) x 128 queries; 8 waves x 16 queries.
// S^T = K*Q^T so softmax'd score tiles feed P*V A-operand with no shuffles.
// ---------------------------------------------------------------------------
static __device__ __forceinline__ void stage_kv(const _Float16* Kbase, const _Float16* Vbase,
                                                _Float16* kbuf, _Float16* vbuf,
                                                int k0, int tid) {
#pragma unroll
  for (int j = 0; j < 2; j++) {  // K tile: 64 keys x 64 d
    int c = tid + j * 256;
    int row = c >> 3, col = (c & 7) << 3;
    async_cp16(Kbase + (size_t)(k0 + row) * HDIM + col, kbuf + row * 72 + col);
  }
#pragma unroll
  for (int j = 0; j < 2; j++) {  // V^T tile: 64 d x 64 keys
    int c = tid + j * 256;
    int row = c >> 3, col = (c & 7) << 3;
    async_cp16(Vbase + (size_t)row * SEQ + k0 + col, vbuf + row * 72 + col);
  }
}

__global__ __launch_bounds__(256) void attn_kernel(const _Float16* __restrict__ Qh,
                                                   const _Float16* __restrict__ Kh,
                                                   const _Float16* __restrict__ Vt,
                                                   const float* __restrict__ mask,
                                                   float* __restrict__ out) {
  constexpr int KSTR = 72;  // 64 + 8 pad halves
  __shared__ alignas(16) _Float16 ks[2][64 * KSTR];
  __shared__ alignas(16) _Float16 vs[2][64 * KSTR];
  __shared__ alignas(16) float ms[SEQ];

  int bh = blockIdx.y, b = bh >> 4, h = bh & 15;
  int tid = threadIdx.x, lane = tid & 31, w = tid >> 5;
  int q0 = blockIdx.x * 128 + w * 16;
  int roff = (lane >> 4) << 3;  // 0 or 8
  int dsel = (lane >> 4) << 4;  // 0 or 16

  const _Float16* Qbase = Qh + (size_t)bh * SEQ * HDIM;
  const _Float16* Kbase = Kh + (size_t)bh * SEQ * HDIM;
  const _Float16* Vbase = Vt + (size_t)bh * HDIM * SEQ;

  {  // hoist the additive mask row for this b into LDS (once per block)
    const float* mp = mask + (size_t)b * SEQ + tid * 8;
    *(v4f*)(ms + tid * 8) = *(const v4f*)mp;
    *(v4f*)(ms + tid * 8 + 4) = *(const v4f*)(mp + 4);
  }

  // Q B-operands (row = query, d halves 8*(lane/16)+{0..7}, 16+8*(lane/16)+{0..7}), reused for all 32 KV steps
  const _Float16* qp = Qbase + (size_t)(q0 + (lane & 15)) * HDIM;
  v16h Qop0 = cat8(*(const v8h*)(qp + roff), *(const v8h*)(qp + roff + 16));
  v16h Qop1 = cat8(*(const v8h*)(qp + 32 + roff), *(const v8h*)(qp + 32 + roff + 16));

  v8f z8 = {0.f, 0.f, 0.f, 0.f, 0.f, 0.f, 0.f, 0.f};
  v8f acc[4] = {z8, z8, z8, z8};
  float m_i = -1e30f, l_i = 0.f;

  stage_kv(Kbase, Vbase, ks[0], vs[0], 0, tid);

  for (int k0 = 0; k0 < SEQ; k0 += 64) {
    int cur = (k0 >> 6) & 1;
    bool haveNext = (k0 + 64) < SEQ;
    if (haveNext) stage_kv(Kbase, Vbase, ks[cur ^ 1], vs[cur ^ 1], k0 + 64, tid);
#if HAVE_ASYNC_LDS
    if (haveNext) __builtin_amdgcn_s_wait_asynccnt(4);  // previous batch landed
    else          __builtin_amdgcn_s_wait_asynccnt(0);
#endif
    __syncthreads();

    const _Float16* kb = ks[cur];
    const _Float16* vb = vs[cur];

    // hoist all 8 K A-operand builds into distinct registers so the
    // ds_load_b128s pipeline instead of serializing on one register octet
    v16h Ak[4][2];
#pragma unroll
    for (int t = 0; t < 4; t++) {
      Ak[t][0] = opA(kb, KSTR, t * 16, 0, lane);
      Ak[t][1] = opA(kb, KSTR, t * 16, 32, lane);
    }
    // scores: 4 S^T tiles (M=key, N=query), contraction over d (2 x k=32)
    v8f sc[4];
#pragma unroll
    for (int t = 0; t < 4; t++) {
      sc[t] = WMMA(Ak[t][0], Qop0, z8);
      sc[t] = WMMA(Ak[t][1], Qop1, sc[t]);
    }

    // issue V B-operand loads now: independent of softmax, their LDS latency
    // hides under the exp/max VALU work below
    v16h Vb[4][2];
#pragma unroll
    for (int dt = 0; dt < 4; dt++) {
      Vb[dt][0] = opB(vb, KSTR, dt * 16, 0, lane);
      Vb[dt][1] = opB(vb, KSTR, dt * 16, 32, lane);
    }

    // additive mask: lane value r of tile t -> key k0 + t*16 + roff + r
#pragma unroll
    for (int t = 0; t < 4; t++) {
      v4f ma = *(const v4f*)(ms + k0 + t * 16 + roff);
      v4f mb = *(const v4f*)(ms + k0 + t * 16 + roff + 4);
#pragma unroll
      for (int r = 0; r < 4; r++) { sc[t][r] += ma[r]; sc[t][r + 4] += mb[r]; }
    }

    // online softmax over 64 keys: 32 in-lane values + partner lane (xor 16)
    float mx = -1e30f;
#pragma unroll
    for (int t = 0; t < 4; t++)
#pragma unroll
      for (int r = 0; r < 8; r++) mx = fmaxf(mx, sc[t][r]);
    mx = fmaxf(mx, __shfl_xor(mx, 16, 32));
    float m_new = fmaxf(m_i, mx);

    float p[4][8], lsum = 0.f;
#pragma unroll
    for (int t = 0; t < 4; t++)
#pragma unroll
      for (int r = 0; r < 8; r++) { p[t][r] = __expf(sc[t][r] - m_new); lsum += p[t][r]; }
    lsum += __shfl_xor(lsum, 16, 32);
    float corr = __expf(m_i - m_new);
    l_i = l_i * corr + lsum;
    m_i = m_new;
#pragma unroll
    for (int t = 0; t < 4; t++) acc[t] *= corr;

    // P as A-operands: Pa = keys [0,32), Pb = keys [32,64) of this step
    union { v16h v; _Float16 e[16]; } Pa, Pb;
#pragma unroll
    for (int r = 0; r < 8; r++) {
      Pa.e[r] = (_Float16)p[0][r]; Pa.e[8 + r] = (_Float16)p[1][r];
      Pb.e[r] = (_Float16)p[2][r]; Pb.e[8 + r] = (_Float16)p[3][r];
    }
#pragma unroll
    for (int dt = 0; dt < 4; dt++) acc[dt] = WMMA(Pa.v, Vb[dt][0], acc[dt]);
#pragma unroll
    for (int dt = 0; dt < 4; dt++) acc[dt] = WMMA(Pb.v, Vb[dt][1], acc[dt]);
    __syncthreads();
  }

  // epilogue: ctx = acc / l. D-tile: lane = d, r = query.
  float inv = 1.0f / l_i;
  int qbase = q0 + roff;
#pragma unroll
  for (int dt = 0; dt < 4; dt++) {
    int d = dt * 16 + (lane & 15);
#pragma unroll
    for (int r = 0; r < 8; r++) {
      out[((size_t)b * SEQ + qbase + r) * HIDDEN + h * HDIM + d] = acc[dt][r] * inv;
    }
  }
}

// ---------------------------------------------------------------------------
extern "C" void kernel_launch(void* const* d_in, const int* in_sizes, int n_in,
                              void* d_out, int out_size, void* d_ws, size_t ws_size,
                              hipStream_t stream) {
  (void)in_sizes; (void)n_in; (void)out_size; (void)ws_size;
  const float* x    = (const float*)d_in[0];
  const float* mask = (const float*)d_in[1];
  const float* Wq   = (const float*)d_in[2];
  const float* bq   = (const float*)d_in[3];
  const float* Wk   = (const float*)d_in[4];
  const float* bk   = (const float*)d_in[5];
  const float* Wv   = (const float*)d_in[6];
  const float* bv   = (const float*)d_in[7];
  float* out = (float*)d_out;

  // workspace carve-up (halves): Xh 8M | Wt 3M | Qh 8M | Kh 8M | Vt 8M = 70 MB
  _Float16* Xh = (_Float16*)d_ws;
  _Float16* Wt = Xh + (size_t)MTOT * HIDDEN;
  _Float16* Qh = Wt + (size_t)3 * HIDDEN * HIDDEN;
  _Float16* Kh = Qh + (size_t)MTOT * HIDDEN;
  _Float16* Vt = Kh + (size_t)MTOT * HIDDEN;

  cvt_x_kernel<<<(MTOT * HIDDEN) / (256 * 8), 256, 0, stream>>>(x, Xh);
  cvt_wt_kernel<<<dim3(HIDDEN / 32, HIDDEN / 32, 3), 256, 0, stream>>>(Wq, Wk, Wv, Wt);
  dim3 pgrid(HIDDEN / 128, MTOT / 256);
  proj_kernel<true><<<pgrid, 256, 0, stream>>>(Xh, Wt, bq, Qh, 0.125f);
  proj_kernel<true><<<pgrid, 256, 0, stream>>>(Xh, Wt + (size_t)HIDDEN * HIDDEN, bk, Kh, 1.0f);
  proj_kernel<false><<<pgrid, 256, 0, stream>>>(Xh, Wt + (size_t)2 * HIDDEN * HIDDEN, bv, Vt, 1.0f);
  attn_kernel<<<dim3(SEQ / 128, NB * HEADS), 256, 0, stream>>>(Qh, Kh, Vt, mask, out);
}
